// ConvexUpsampler_11218454577862
// MI455X (gfx1250) — compile-verified, matrix operand order checked by reading, not run on hardware
//
#include <hip/hip_runtime.h>

// MI455X / gfx1250, wave32. Convex flow upsampler.
// conv1 (3x3, 256->256) is ~87 GFLOP -> compute bound -> f16 WMMA w/ f32 accum.
// A-tile staging uses gfx1250 async global->LDS copies (ASYNCcnt path),
// double-buffered with 1 barrier/K-step. Needs ~77 MB of d_ws.

typedef _Float16 v8h  __attribute__((ext_vector_type(8)));
typedef _Float16 v16h __attribute__((ext_vector_type(16)));
typedef float    v8f  __attribute__((ext_vector_type(8)));
typedef int      v4i  __attribute__((ext_vector_type(4)));

#define IN_CH 256
#define HGT   96
#define WID   96
#define NB    8
#define HWPIX (HGT*WID)        // 9216
#define NPIX  (NB*HWPIX)       // 73728
#define K1    (9*IN_CH)        // 2304
#define N2    576
#define UPF   8

#define AS1 __attribute__((address_space(1)))
#define AS3 __attribute__((address_space(3)))

#if defined(__has_builtin)
#  if __has_builtin(__builtin_amdgcn_global_load_async_to_lds_b128)
#    define HAVE_ASYNC_LDS 1
#  endif
#  if __has_builtin(__builtin_amdgcn_s_wait_asynccnt)
#    define HAVE_WAIT_ASYNC 1
#  endif
#endif
#ifndef HAVE_ASYNC_LDS
#  define HAVE_ASYNC_LDS 0
#endif
#ifndef HAVE_WAIT_ASYNC
#  define HAVE_WAIT_ASYNC 0
#endif

static __device__ __forceinline__ v16h cat16(v8h lo, v8h hi) {
  return __builtin_shufflevector(lo, hi, 0,1,2,3,4,5,6,7,8,9,10,11,12,13,14,15);
}

// -------- feat NCHW f32 -> NHWC f16 (LDS-tiled transpose, coalesced both sides)
__global__ __launch_bounds__(256) void nchw_to_nhwc_f16(const float* __restrict__ in,
                                                        _Float16* __restrict__ out) {
  __shared__ float tile[32][33];
  const int bx = blockIdx.x;            // hw tile (9216/32 = 288)
  const int by = blockIdx.y;            // c tile  (256/32  = 8)
  const int b  = blockIdx.z;
  const int tx = threadIdx.x, ty = threadIdx.y;
#pragma unroll
  for (int i = 0; i < 32; i += 8) {
    const int c  = by*32 + ty + i;
    const int hw = bx*32 + tx;
    tile[ty + i][tx] = in[(size_t)(b*IN_CH + c)*HWPIX + hw];
  }
  __syncthreads();
#pragma unroll
  for (int i = 0; i < 32; i += 8) {
    const int hw = bx*32 + ty + i;
    const int c  = by*32 + tx;
    out[(size_t)(b*HWPIX + hw)*IN_CH + c] = (_Float16)tile[tx][ty + i];
  }
}

// -------- w1 (co,ci,3,3) f32 -> w1h[co][r*256+ci] f16  (K laid out r-major)
__global__ void reorder_w1(const float* __restrict__ w1, _Float16* __restrict__ w1h) {
  const int idx = blockIdx.x*blockDim.x + threadIdx.x;
  if (idx >= IN_CH*K1) return;
  const int co = idx / K1;
  const int k  = idx % K1;
  const int r  = k / IN_CH;
  const int ci = k % IN_CH;
  w1h[idx] = (_Float16)w1[(co*IN_CH + ci)*9 + r];
}

// -------- w2 (576,256) f32 -> f16 (layout already [m][ci])
__global__ void convert_w2(const float* __restrict__ w2, _Float16* __restrict__ w2h) {
  const int idx = blockIdx.x*blockDim.x + threadIdx.x;
  if (idx < N2*IN_CH) w2h[idx] = (_Float16)w2[idx];
}

// -------- GEMM1: implicit-GEMM 3x3 conv. M=73728, N=256, K=2304.
// Workgroup 256 thr = 8 waves; tile 128x128; wave tile 32x64 -> 8 wmma accs.
// Double-buffered A tile in LDS (async global->LDS when available, 1 barrier
// per K-step); B fragments direct from L2-resident repacked weights.
__global__ __launch_bounds__(256) void conv3x3_wmma(const _Float16* __restrict__ featH,
                                                    const _Float16* __restrict__ w1h,
                                                    const float* __restrict__ b1,
                                                    _Float16* __restrict__ xh) {
  __shared__ __align__(16) _Float16 As[2][128*40];  // 2 x 10 KB, stride 40 halves
  const int tid  = threadIdx.x;
  const int lane = tid & 31;
  const int wave = tid >> 5;
  const int wm   = wave & 3;                     // 4 waves along M
  const int wn   = wave >> 2;                    // 2 waves along N
  const int m0   = blockIdx.x * 128;
  const int n0   = blockIdx.y * 128;

  // A-tile fill mapping: thread -> (row, 16-half chunk)
  const int arow  = tid >> 1;
  const int apart = tid & 1;
  const int am   = m0 + arow;
  const int ab   = am / HWPIX;
  const int arem = am % HWPIX;
  const int ah   = arem / WID;
  const int aw   = arem % WID;

  // per-lane fragment coords (16-bit WMMA operand layout)
  const int fr = lane & 15;                      // M row (A) / N col (B)
  const int kh = (lane >> 4) * 8;                // K chunk base: {0,8}, +16 for hi half

  v8f acc[2][4] = {};

  // stage one 128x32 A slice (K-step `step`) into buffer `buf`
  auto stage = [&](int step, int buf) {
    const int r  = step >> 3;
    const int kc = step & 7;
    const int hh = ah + (r / 3) - 1;
    const int ww = aw + (r % 3) - 1;
    const bool valid = ((unsigned)hh < (unsigned)HGT) && ((unsigned)ww < (unsigned)WID);
    _Float16* dst = &As[buf][arow*40 + apart*16];
    const _Float16* src = featH + (size_t)(ab*HWPIX + hh*WID + ww)*IN_CH + kc*32 + apart*16;
#if HAVE_ASYNC_LDS
    if (valid) {
      __builtin_amdgcn_global_load_async_to_lds_b128((AS1 v4i*)src,
                                                     (AS3 v4i*)dst, 0, 0);
      __builtin_amdgcn_global_load_async_to_lds_b128((AS1 v4i*)(src + 8),
                                                     (AS3 v4i*)(dst + 8), 0, 0);
    } else {
      v8h z = {};
      *(v8h*)dst       = z;
      *(v8h*)(dst + 8) = z;
    }
#else
    v8h a0 = {}, a1 = {};
    if (valid) {
      a0 = *(const v8h*)src;
      a1 = *(const v8h*)(src + 8);
    }
    *(v8h*)dst       = a0;
    *(v8h*)(dst + 8) = a1;
#endif
  };

  auto commit = [&]() {
#if HAVE_ASYNC_LDS && HAVE_WAIT_ASYNC
    __builtin_amdgcn_s_wait_asynccnt(0);   // own async LDS writes done
#endif
    __syncthreads();                        // covers DScnt + makes writes visible
  };

  stage(0, 0);
  commit();

  for (int step = 0; step < 72; ++step) {
    const int buf = step & 1;
    if (step + 1 < 72) stage(step + 1, buf ^ 1);   // overlap next slice with WMMAs

    const int r   = step >> 3;
    const int ci0 = (step & 7) * 32;

    v16h bf[4];
#pragma unroll
    for (int nt = 0; nt < 4; ++nt) {
      const int n = n0 + wn*64 + nt*16 + fr;
      const _Float16* bp = w1h + (size_t)n*K1 + r*IN_CH + ci0 + kh;
      bf[nt] = cat16(*(const v8h*)bp, *(const v8h*)(bp + 16));
    }
    v16h af[2];
#pragma unroll
    for (int mt = 0; mt < 2; ++mt) {
      const int row = wm*32 + mt*16 + fr;
      af[mt] = cat16(*(const v8h*)&As[buf][row*40 + kh],
                     *(const v8h*)&As[buf][row*40 + kh + 16]);
    }
#pragma unroll
    for (int mt = 0; mt < 2; ++mt)
#pragma unroll
      for (int nt = 0; nt < 4; ++nt)
        acc[mt][nt] = __builtin_amdgcn_wmma_f32_16x16x32_f16(
            false, af[mt], false, bf[nt], (short)0, acc[mt][nt], false, false);

    commit();
  }

  // epilogue: +bias, ReLU, store x as NHWC f16
#pragma unroll
  for (int nt = 0; nt < 4; ++nt) {
    const int n = n0 + wn*64 + nt*16 + fr;
    const float bias = b1[n];
#pragma unroll
    for (int mt = 0; mt < 2; ++mt)
#pragma unroll
      for (int j = 0; j < 8; ++j) {
        const int m = m0 + wm*32 + mt*16 + (lane>>4)*8 + j;
        float v = acc[mt][nt][j] + bias;
        v = v > 0.f ? v : 0.f;
        xh[(size_t)m*IN_CH + n] = (_Float16)v;
      }
  }
}

// -------- GEMM2 fused: 1x1 conv (N=576) + softmax(9) + convex combine + shuffle.
// 9 waves, wave k owns neighbor k's 64 (p,q) channels (c = k*64 + p*8 + q).
// 32 pixels / workgroup; logits parked in LDS; mask never hits HBM.
__global__ __launch_bounds__(288) void mask_softmax_combine(const _Float16* __restrict__ xh,
                                                            const _Float16* __restrict__ w2h,
                                                            const float* __restrict__ b2,
                                                            const float* __restrict__ flow,
                                                            float* __restrict__ out) {
  __shared__ __align__(16) _Float16 As[32*40];   // 2.5 KB
  __shared__ float maskLds[9*32*64];             // 72 KB: [k][pixel][pq]
  const int tid  = threadIdx.x;
  const int lane = tid & 31;
  const int wave = tid >> 5;                     // 0..8 == neighbor index k
  const int m0   = blockIdx.x * 32;
  const int fr   = lane & 15;
  const int kh   = (lane >> 4) * 8;

  v8f acc[2][4] = {};

  for (int ks = 0; ks < 8; ++ks) {
    const int k0 = ks*32;
    if (tid < 64) {
      const int row = tid >> 1, part = tid & 1;
      const _Float16* src = xh + (size_t)(m0 + row)*IN_CH + k0 + part*16;
      *(v8h*)&As[row*40 + part*16]     = *(const v8h*)src;
      *(v8h*)&As[row*40 + part*16 + 8] = *(const v8h*)(src + 8);
    }
    __syncthreads();

    v16h bf[4];
#pragma unroll
    for (int nt = 0; nt < 4; ++nt) {
      const int n = wave*64 + nt*16 + fr;
      const _Float16* bp = w2h + (size_t)n*IN_CH + k0 + kh;
      bf[nt] = cat16(*(const v8h*)bp, *(const v8h*)(bp + 16));
    }
    v16h af[2];
#pragma unroll
    for (int mt = 0; mt < 2; ++mt) {
      const int row = mt*16 + fr;
      af[mt] = cat16(*(const v8h*)&As[row*40 + kh],
                     *(const v8h*)&As[row*40 + kh + 16]);
    }
#pragma unroll
    for (int mt = 0; mt < 2; ++mt)
#pragma unroll
      for (int nt = 0; nt < 4; ++nt)
        acc[mt][nt] = __builtin_amdgcn_wmma_f32_16x16x32_f16(
            false, af[mt], false, bf[nt], (short)0, acc[mt][nt], false, false);
    __syncthreads();
  }

  // park logits (+bias) in LDS
#pragma unroll
  for (int nt = 0; nt < 4; ++nt) {
    const float bias = b2[wave*64 + nt*16 + fr];
#pragma unroll
    for (int mt = 0; mt < 2; ++mt)
#pragma unroll
      for (int j = 0; j < 8; ++j) {
        const int p_local = mt*16 + (lane>>4)*8 + j;
        const int pq      = nt*16 + fr;
        maskLds[(wave*32 + p_local)*64 + pq] = acc[mt][nt][j] + bias;
      }
  }
  __syncthreads();

  // cross-wave softmax over the 9 neighbors + convex combination + pixel shuffle
  for (int idx = tid; idx < 32*64; idx += 288) {
    const int p_local = idx >> 6;
    const int pq      = idx & 63;
    const int m   = m0 + p_local;
    const int b   = m / HWPIX;
    const int rem = m % HWPIX;
    const int h   = rem / WID;
    const int w   = rem % WID;
    float v[9], mx = -3.0e38f;
#pragma unroll
    for (int k = 0; k < 9; ++k) {
      v[k] = maskLds[(k*32 + p_local)*64 + pq];
      mx = fmaxf(mx, v[k]);
    }
    float s0 = 0.f, s1 = 0.f, wsum = 0.f;
#pragma unroll
    for (int k = 0; k < 9; ++k) {
      const float e = __expf(v[k] - mx);
      wsum += e;
      const int hh = h + k/3 - 1, ww = w + k%3 - 1;
      if (hh >= 0 && hh < HGT && ww >= 0 && ww < WID) {
        const size_t base = ((size_t)(b*2)*HGT + hh)*WID + ww;
        s0 += e * flow[base];
        s1 += e * flow[base + (size_t)HGT*WID];
      }
    }
    const float scl = (float)UPF / wsum;
    const int p = pq >> 3, q = pq & 7;
    const int oh = h*UPF + p, ow = w*UPF + q;
    out[((size_t)(b*2 + 0)*(HGT*UPF) + oh)*(WID*UPF) + ow] = s0 * scl;
    out[((size_t)(b*2 + 1)*(HGT*UPF) + oh)*(WID*UPF) + ow] = s1 * scl;
  }
}

extern "C" void kernel_launch(void* const* d_in, const int* in_sizes, int n_in,
                              void* d_out, int out_size, void* d_ws, size_t ws_size,
                              hipStream_t stream) {
  const float* flow = (const float*)d_in[0];
  const float* feat = (const float*)d_in[1];
  const float* w1   = (const float*)d_in[2];
  const float* b1   = (const float*)d_in[3];
  const float* w2   = (const float*)d_in[4];
  const float* b2   = (const float*)d_in[5];
  float* out = (float*)d_out;

  // workspace layout (f16): featH | xh | w1h | w2h  (~77 MB total)
  _Float16* featH = (_Float16*)d_ws;
  _Float16* xh    = featH + (size_t)NPIX*IN_CH;
  _Float16* w1h   = xh    + (size_t)NPIX*IN_CH;
  _Float16* w2h   = w1h   + (size_t)IN_CH*K1;

  nchw_to_nhwc_f16<<<dim3(HWPIX/32, IN_CH/32, NB), dim3(32, 8), 0, stream>>>(feat, featH);
  reorder_w1<<<(IN_CH*K1 + 255)/256, 256, 0, stream>>>(w1, w1h);
  convert_w2<<<(N2*IN_CH + 255)/256, 256, 0, stream>>>(w2, w2h);
  conv3x3_wmma<<<dim3(NPIX/128, IN_CH/128), 256, 0, stream>>>(featH, w1h, b1, xh);
  mask_softmax_combine<<<NPIX/32, 288, 0, stream>>>(xh, w2h, b2, flow, out);
}
